// GlobalAttentionPool_52716428591482
// MI455X (gfx1250) — matrix operand order repaired and checked by reading, not compiled, not run on hardware
//
#include <hip/hip_runtime.h>

// ---------------------------------------------------------------------------
// GlobalAttentionPool for MI455X (gfx1250, wave32, WMMA)
//   beta = tanh([x | x_root] @ W1 + b1) @ W2        (bf16 WMMA, f32 accum)
//   out[b] = sum_n softmax_seg(beta)[n] * x[n]
// ---------------------------------------------------------------------------

typedef __bf16 bf16_t;
typedef bf16_t v16bf __attribute__((ext_vector_type(16)));
typedef float  v8f   __attribute__((ext_vector_type(8)));

#define FDIM  128
#define HDIM  128
#define TWOF  256
#define APAD  8      // LDS row padding (bf16 elems) to avoid bank conflicts
#define MROWS 96     // nodes per block (6 M-tiles of 16); LDS ~50KB
#define MTILES (MROWS / 16)

// Fast device math: native TRANS ops instead of branchy libm expansions.
__device__ __forceinline__ float fast_tanh(float x)
{
#if __has_builtin(__builtin_amdgcn_tanhf)
    return __builtin_amdgcn_tanhf(x);          // v_tanh_f32 (CDNA5 native)
#else
    float e = __expf(2.0f * x);                // v_exp_f32 path
    return 1.0f - 2.0f * __builtin_amdgcn_rcpf(e + 1.0f);
#endif
}

// ---------------------------------------------------------------------------
// Kernel 0: first[b] = lower_bound(batch, b) for b in [0, B]  (first[B] == N)
// ---------------------------------------------------------------------------
__global__ void seg_start_kernel(const long long* __restrict__ batch,
                                 int* __restrict__ first, int N, int B)
{
    int b = blockIdx.x * blockDim.x + threadIdx.x;
    if (b > B) return;
    long long key = (long long)b;
    int lo = 0, hi = N;
    while (lo < hi) {
        int mid = (lo + hi) >> 1;
        if (batch[mid] < key) lo = mid + 1; else hi = mid;
    }
    first[b] = lo;
}

// ---------------------------------------------------------------------------
// Kernel 0b: pre-pack W1 (f32, [256,128] row-major) into bf16 laid out as
// WMMA B-fragments: Wp[((wave*8 + kk)*32 + lane)*16 + j]
//   where for (K, ncol): wave=ncol/16, mrow=ncol%16, kk=K/32,
//                        hi=(K%32)/16, j=K%16, lane=hi*16+mrow.
// Each lane of the main kernel then loads one v16bf (32B) per fragment.
// ---------------------------------------------------------------------------
__global__ void pack_w1_kernel(const float* __restrict__ W1,
                               bf16_t* __restrict__ Wp)
{
    int i = blockIdx.x * blockDim.x + threadIdx.x;   // 0 .. 256*128-1
    if (i >= TWOF * HDIM) return;
    int K    = i / HDIM;
    int ncol = i % HDIM;
    int w    = ncol >> 4;
    int mrow = ncol & 15;
    int kk   = K >> 5;
    int rem  = K & 31;
    int hi   = rem >> 4;
    int j    = rem & 15;
    int lane = hi * 16 + mrow;
    Wp[(((w * 8 + kk) * 32) + lane) * 16 + j] = (bf16_t)W1[i];
}

// ---------------------------------------------------------------------------
// Kernel 1: attention scores via WMMA.
// Block = 256 threads = 8 waves; handles MROWS=96 nodes (6 M-tiles).
// Wave w owns hidden columns [16w, 16w+16); B fragments (8 K-steps) are
// hoisted into 64 VGPRs and reused across the 6 M-tiles (48 WMMAs / wave).
// Cross-lane + cross-wave reduction of beta partials is done entirely by the
// LDS atomic unit (ds_add_f32, no return) -- no bpermute index math.
// ---------------------------------------------------------------------------
__global__ __launch_bounds__(256)
void attn_score_kernel(const float* __restrict__ x,
                       const long long* __restrict__ batch,
                       const int* __restrict__ first,
                       const bf16_t* __restrict__ Wp,  // packed B fragments
                       const float* __restrict__ b1,   // [128]
                       const float* __restrict__ W2,   // [128]
                       float* __restrict__ beta, int N)
{
    __shared__ bf16_t sA[MROWS][TWOF + APAD];
    __shared__ float  sBeta[MROWS];
    __shared__ int    sRoot[MROWS];

    const int tid   = threadIdx.x;
    const int node0 = blockIdx.x * MROWS;

    // ---- Stage root indices once per block --------------------------------
    if (tid < MROWS) {
        int node = node0 + tid;
        int cl   = node < N ? node : N - 1;
        sRoot[tid] = first[(int)batch[cl]];
        sBeta[tid] = 0.0f;
    }
    __syncthreads();

    // ---- Stage A-tile into LDS: MROWS x 256 cols ([x | x_root]) bf16 ------
    {
        const int seg = tid & 15;        // 0..15  (8-float chunk)
        #pragma unroll
        for (int mt = 0; mt < MTILES; ++mt) {
            int row  = mt * 16 + (tid >> 4);
            int node = node0 + row;
            int cl   = node < N ? node : N - 1;
            int root = sRoot[row];

            const float4* xv = (const float4*)(x + (size_t)cl   * FDIM) + seg * 2;
            const float4* rv = (const float4*)(x + (size_t)root * FDIM) + seg * 2;
            float4 a0 = xv[0], a1 = xv[1];
            float4 r0 = rv[0], r1 = rv[1];

            int c0 = seg * 8;
            sA[row][c0 + 0] = (bf16_t)a0.x;  sA[row][c0 + 1] = (bf16_t)a0.y;
            sA[row][c0 + 2] = (bf16_t)a0.z;  sA[row][c0 + 3] = (bf16_t)a0.w;
            sA[row][c0 + 4] = (bf16_t)a1.x;  sA[row][c0 + 5] = (bf16_t)a1.y;
            sA[row][c0 + 6] = (bf16_t)a1.z;  sA[row][c0 + 7] = (bf16_t)a1.w;
            sA[row][FDIM + c0 + 0] = (bf16_t)r0.x;
            sA[row][FDIM + c0 + 1] = (bf16_t)r0.y;
            sA[row][FDIM + c0 + 2] = (bf16_t)r0.z;
            sA[row][FDIM + c0 + 3] = (bf16_t)r0.w;
            sA[row][FDIM + c0 + 4] = (bf16_t)r1.x;
            sA[row][FDIM + c0 + 5] = (bf16_t)r1.y;
            sA[row][FDIM + c0 + 6] = (bf16_t)r1.z;
            sA[row][FDIM + c0 + 7] = (bf16_t)r1.w;
        }
    }
    __syncthreads();

    // ---- Load this wave's B fragments (all 8 K-steps) into registers ------
    const int lane = tid & 31;
    const int wave = tid >> 5;       // n-tile index, 0..7
    const int mrow = lane & 15;      // A: M index; B/C/D: N index
    const int hi   = lane >> 4;      // lane half
    const int ncol = wave * 16 + mrow;

    const v16bf* WpV = (const v16bf*)Wp;     // 32B-aligned fragments
    v16bf bfrag[8];
    #pragma unroll
    for (int kk = 0; kk < 8; ++kk)
        bfrag[kk] = WpV[(wave * 8 + kk) * 32 + lane];

    const float bias = b1[ncol];
    const float w2   = W2[ncol];

    // ---- MTILES x 8 K-steps of v_wmma_f32_16x16x32_bf16 -------------------
    #pragma unroll
    for (int mt = 0; mt < MTILES; ++mt) {
        const int arow = mt * 16 + mrow;
        v8f c = {};
        #pragma unroll
        for (int kk = 0; kk < 8; ++kk) {
            const int kb = kk * 32;
            v16bf a;
            // A fragment (contiguous K order, matching packed B):
            //   hi=0: elems 0-15 -> K=kb+0..15
            //   hi=1: elems 0-15 -> K=kb+16..31
            #pragma unroll
            for (int j = 0; j < 8; ++j) {
                a[j]     = sA[arow][kb + hi * 16 + j];
                a[j + 8] = sA[arow][kb + hi * 16 + 8 + j];
            }
            c = __builtin_amdgcn_wmma_f32_16x16x32_bf16(
                    /*neg_a=*/false, a, /*neg_b=*/false, bfrag[kk],
                    /*c_mod=*/(short)0, c, /*reuse_a=*/false, /*reuse_b=*/false);
        }

        // beta partials: tanh(h + b1) * W2; the LDS atomic unit reduces over
        // lanes (N) and waves.  C/D layout: VGPR r holds M = r + 8*hi, N=mrow.
        #pragma unroll
        for (int r = 0; r < 8; ++r) {
            float v = fast_tanh(c[r] + bias) * w2;     // v_tanh_f32
            atomicAdd(&sBeta[mt * 16 + r + hi * 8], v); // ds_add_f32 (no rtn)
        }
    }
    __syncthreads();

    if (tid < MROWS) {
        int node = node0 + tid;
        if (node < N) beta[node] = sBeta[tid];
    }
}

// ---------------------------------------------------------------------------
// Kernel 2: segment softmax + weighted pooling. One 128-thread block / graph.
// ---------------------------------------------------------------------------
__global__ __launch_bounds__(128)
void pool_kernel(const float* __restrict__ x,
                 const float* __restrict__ beta,
                 const int* __restrict__ first,
                 float* __restrict__ out, int N)
{
    const int b   = blockIdx.x;
    const int tid = threadIdx.x;            // feature index, 0..127
    const int start = first[b];
    const int end   = first[b + 1];

    if (start >= end) {                      // empty graph -> zeros
        out[(size_t)b * FDIM + tid] = 0.0f;
        return;
    }

    __shared__ float red[4];
    const int lane = tid & 31;
    const int wave = tid >> 5;

    // segment max
    float mx = -3.402823466e38f;
    for (int n = start + tid; n < end; n += 128) mx = fmaxf(mx, beta[n]);
    #pragma unroll
    for (int m = 16; m >= 1; m >>= 1) mx = fmaxf(mx, __shfl_xor(mx, m));
    if (lane == 0) red[wave] = mx;
    __syncthreads();
    mx = fmaxf(fmaxf(red[0], red[1]), fmaxf(red[2], red[3]));
    __syncthreads();

    // segment sum of exp (native v_exp_f32)
    float s = 0.0f;
    for (int n = start + tid; n < end; n += 128) s += __expf(beta[n] - mx);
    #pragma unroll
    for (int m = 16; m >= 1; m >>= 1) s += __shfl_xor(s, m);
    if (lane == 0) red[wave] = s;
    __syncthreads();
    s = red[0] + red[1] + red[2] + red[3];
    const float inv_s = __builtin_amdgcn_rcpf(s);   // v_rcp_f32

    // weighted pooling: thread tid owns feature tid (coalesced over nodes)
    float acc = 0.0f;
    for (int n = start; n < end; ++n)
        acc += __expf(beta[n] - mx) * x[(size_t)n * FDIM + tid];
    out[(size_t)b * FDIM + tid] = acc * inv_s;
}

// ---------------------------------------------------------------------------
// Host entry
// ---------------------------------------------------------------------------
extern "C" void kernel_launch(void* const* d_in, const int* in_sizes, int n_in,
                              void* d_out, int out_size, void* d_ws, size_t ws_size,
                              hipStream_t stream)
{
    const float*     x     = (const float*)d_in[0];
    const long long* batch = (const long long*)d_in[1];
    const float*     W1    = (const float*)d_in[2];
    const float*     b1    = (const float*)d_in[3];
    const float*     W2    = (const float*)d_in[4];
    float*           out   = (float*)d_out;

    const int N = in_sizes[1];
    const int B = out_size / FDIM;

    // workspace layout: first[B+1] ints | Wp[256*128] bf16 | beta[N] floats
    int*    first = (int*)d_ws;
    size_t  off1  = (((size_t)(B + 1) * sizeof(int)) + 255) & ~(size_t)255;
    bf16_t* Wp    = (bf16_t*)((char*)d_ws + off1);
    size_t  off2  = (off1 + (size_t)TWOF * HDIM * sizeof(bf16_t) + 255) & ~(size_t)255;
    float*  beta  = (float*)((char*)d_ws + off2);

    seg_start_kernel<<<(B + 1 + 255) / 256, 256, 0, stream>>>(batch, first, N, B);

    pack_w1_kernel<<<(TWOF * HDIM + 255) / 256, 256, 0, stream>>>(W1, Wp);

    const int mblocks = (N + MROWS - 1) / MROWS;
    attn_score_kernel<<<mblocks, 256, 0, stream>>>(x, batch, first, Wp, b1, W2,
                                                   beta, N);

    pool_kernel<<<B, 128, 0, stream>>>(x, beta, first, out, N);
}
